// HyperNet_644245094419
// MI455X (gfx1250) — matrix operand order repaired and checked by reading, not compiled, not run on hardware
//
#include <hip/hip_runtime.h>

typedef __attribute__((ext_vector_type(16))) _Float16 v16h;
typedef __attribute__((ext_vector_type(8)))  _Float16 v8h;
typedef __attribute__((ext_vector_type(8)))  float    v8f;
typedef __attribute__((ext_vector_type(4)))  int      v4i;

#define AS1 __attribute__((address_space(1)))
#define AS3 __attribute__((address_space(3)))

#if __has_builtin(__builtin_amdgcn_global_load_async_to_lds_b128)
#define USE_ASYNC_LDS 1
#else
#define USE_ASYNC_LDS 0
#endif

static __device__ __forceinline__ void wait_async0() {
#if __has_builtin(__builtin_amdgcn_s_wait_asynccnt)
  __builtin_amdgcn_s_wait_asynccnt(0);
#else
  asm volatile("s_wait_asynccnt 0x0" ::: "memory");
#endif
}

static __device__ __forceinline__ v8f wmma_f16(v16h a, v16h b, v8f c) {
  // D = A(16x32 f16) * B(32x16 f16) + C(16x16 f32)
  return __builtin_amdgcn_wmma_f32_16x16x32_f16(false, a, false, b, (short)0, c,
                                                false, false);
}

static __device__ __forceinline__ v16h cat8(v8h lo, v8h hi) {
  return __builtin_shufflevector(lo, hi, 0, 1, 2, 3, 4, 5, 6, 7,
                                 8, 9, 10, 11, 12, 13, 14, 15);
}

// ---------------------------------------------------------------------------
// Transpose-and-cast: src f32 [R,C] row-major -> dst f16 [C,R] row-major (B^T)
// ---------------------------------------------------------------------------
__global__ void k_cvt_t(const float* __restrict__ src, _Float16* __restrict__ dst,
                        int R, int C) {
  int i = blockIdx.x * 256 + threadIdx.x;
  if (i >= R * C) return;
  int r = i / C, c = i - r * C;
  dst[(size_t)c * R + r] = (_Float16)src[i];
}

// ---------------------------------------------------------------------------
// Generic WMMA GEMM for the small stages.
// Each wave computes FOUR 16x16 M-tiles of one N-tile: the B fragment is
// loaded once per K-step and reused 4x (4x less weight re-read traffic).
// ---------------------------------------------------------------------------
template <bool AF32, bool RELU, bool BIAS>
__global__ __launch_bounds__(256) void k_gemm16x4(
    const void* __restrict__ Aptr, const _Float16* __restrict__ Bt,
    const float* __restrict__ bias, _Float16* __restrict__ C,
    int M, int N, int K, int lda, int ldb, int ldc, int coff) {
  const int NT = N >> 4;
  const int tile = blockIdx.x * 8 + (threadIdx.x >> 5);
  if (tile >= (M >> 6) * NT) return;
  const int m0 = (tile / NT) << 6;  // 64 rows per wave
  const int n0 = (tile % NT) << 4;
  const int lane = threadIdx.x & 31;
  const int lanelo = lane & 15;
  const int hsel = (lane >> 4) & 1;
  const int gn = n0 + lanelo;

  v8f acc[4] = {{}, {}, {}, {}};
  const _Float16* brow = Bt + (size_t)gn * ldb + hsel * 8;

  for (int kk = 0; kk < K; kk += 32) {
    v16h b = cat8(*(const v8h*)(brow + kk), *(const v8h*)(brow + kk + 16));  // shared by the 4 M-tiles
    const int off = kk + hsel * 8;
#pragma unroll
    for (int q = 0; q < 4; ++q) {
      const int gm = m0 + q * 16 + lanelo;
      v8h lo, hi;
      if (AF32) {
        const float* Af = (const float*)Aptr + (size_t)gm * lda + off;
        float4 f0 = *(const float4*)(Af);
        float4 f1 = *(const float4*)(Af + 4);
        float4 f2 = *(const float4*)(Af + 16);
        float4 f3 = *(const float4*)(Af + 20);
        lo = (v8h){(_Float16)f0.x, (_Float16)f0.y, (_Float16)f0.z, (_Float16)f0.w,
                   (_Float16)f1.x, (_Float16)f1.y, (_Float16)f1.z, (_Float16)f1.w};
        hi = (v8h){(_Float16)f2.x, (_Float16)f2.y, (_Float16)f2.z, (_Float16)f2.w,
                   (_Float16)f3.x, (_Float16)f3.y, (_Float16)f3.z, (_Float16)f3.w};
      } else {
        const _Float16* Ah = (const _Float16*)Aptr + (size_t)gm * lda + off;
        lo = *(const v8h*)(Ah);
        hi = *(const v8h*)(Ah + 16);
      }
      acc[q] = wmma_f16(cat8(lo, hi), b, acc[q]);
    }
  }

  float bv = 0.0f;
  if (BIAS) bv = bias[gn];
#pragma unroll
  for (int q = 0; q < 4; ++q) {
    _Float16* Crow = C + (size_t)(m0 + q * 16 + hsel * 8) * ldc + coff + gn;
#pragma unroll
    for (int e = 0; e < 8; ++e) {
      float v = acc[q][e] + bv;
      if (RELU) v = v > 0.0f ? v : 0.0f;
      Crow[(size_t)e * ldc] = (_Float16)v;
    }
  }
}

// ---------------------------------------------------------------------------
// Final GEMM, 2D block tiling: BM=128 x BN=256 per workgroup, 16 waves (4x4),
// each wave 32x64 (2 M-tiles x 4 N-tiles). K chunked by 32 with DOUBLE-
// BUFFERED LDS; staging of chunk k+1 overlaps the WMMAs of chunk k, either
// via GLOBAL_LOAD_ASYNC_TO_LDS_B128 (+ s_wait_asynccnt) when the builtin is
// available, or via register-staged loads issued before compute.
// L2 traffic: B 4MB*128 = 512MB, A 64MB*4 = 256MB (vs 4GB with M=16 tiles).
// Epilogue writes sample and deterministic per-(row, n-block) partial sums.
// ---------------------------------------------------------------------------
__global__ __launch_bounds__(512) void k_final(
    const _Float16* __restrict__ Acat,  // [16384, 2048] f16
    const _Float16* __restrict__ Bt,    // [1024, 2048]  f16 (comb_w^T)
    const float* __restrict__ bias,     // [1024]
    const float* __restrict__ eps,      // [16384, 1024]
    float* __restrict__ sample,         // [16384, 1024]
    float* __restrict__ psum) {         // [16384, 4] partial row sums
  __shared__ alignas(32) _Float16 Asl[2][128 * 32];  // 2 x 8 KB
  __shared__ alignas(32) _Float16 Bsl[2][256 * 32];  // 2 x 16 KB
  __shared__ float wpart[4][32][4];                  // [wm][row][wn]

  const int tid = threadIdx.x;
  const int wave = tid >> 5;
  const int wm = wave >> 2;  // 0..3 : M position
  const int wn = wave & 3;   // 0..3 : N position
  const int lane = tid & 31;
  const int lanelo = lane & 15;
  const int hsel = (lane >> 4) & 1;
  const int m0 = blockIdx.y << 7;  // 128 rows
  const int n0 = blockIdx.x << 8;  // 256 cols

  v8f acc[2][4] = {{{}, {}, {}, {}}, {{}, {}, {}, {}}};

  // Per-thread staging coordinates (A: 512 uint4 = 1/thread; B: 1024 = 2/thr)
  const int sr = tid >> 2;           // A row / B col (first)
  const int sc = (tid & 3) * 8;      // f16 column within 32-wide chunk
  const int sn1 = (tid + 512) >> 2;  // B col (second)

  auto compute = [&](int buf) {
    const int off = hsel * 8;
    v16h a[2];
#pragma unroll
    for (int mt = 0; mt < 2; ++mt) {
      const _Float16* ar = &Asl[buf][(wm * 32 + mt * 16 + lanelo) * 32 + off];
      a[mt] = cat8(*(const v8h*)ar, *(const v8h*)(ar + 16));
    }
#pragma unroll
    for (int t = 0; t < 4; ++t) {
      const _Float16* br =
          &Bsl[buf][(wn * 64 + t * 16 + lanelo) * 32 + hsel * 8];
      v16h b = cat8(*(const v8h*)br, *(const v8h*)(br + 16));
#pragma unroll
      for (int mt = 0; mt < 2; ++mt) acc[mt][t] = wmma_f16(a[mt], b, acc[mt][t]);
    }
  };

#if USE_ASYNC_LDS
  auto stage = [&](int buf, int kc) {
    __builtin_amdgcn_global_load_async_to_lds_b128(
        (AS1 v4i*)(Acat + (size_t)(m0 + sr) * 2048 + kc * 32 + sc),
        (AS3 v4i*)(&Asl[buf][sr * 32 + sc]), 0, 0);
    __builtin_amdgcn_global_load_async_to_lds_b128(
        (AS1 v4i*)(Bt + (size_t)(n0 + sr) * 2048 + kc * 32 + sc),
        (AS3 v4i*)(&Bsl[buf][sr * 32 + sc]), 0, 0);
    __builtin_amdgcn_global_load_async_to_lds_b128(
        (AS1 v4i*)(Bt + (size_t)(n0 + sn1) * 2048 + kc * 32 + sc),
        (AS3 v4i*)(&Bsl[buf][sn1 * 32 + sc]), 0, 0);
  };
  stage(0, 0);
  for (int kc = 0; kc < 64; ++kc) {
    wait_async0();    // this wave's async LDS writes done
    __syncthreads();  // -> whole chunk visible to all waves
    if (kc + 1 < 64) stage((kc + 1) & 1, kc + 1);  // overlap with compute
    compute(kc & 1);
  }
#else
  uint4 ra, rb0, rb1;
  auto gload = [&](int kc) {
    ra  = *(const uint4*)(Acat + (size_t)(m0 + sr) * 2048 + kc * 32 + sc);
    rb0 = *(const uint4*)(Bt + (size_t)(n0 + sr) * 2048 + kc * 32 + sc);
    rb1 = *(const uint4*)(Bt + (size_t)(n0 + sn1) * 2048 + kc * 32 + sc);
  };
  auto sstore = [&](int buf) {
    *(uint4*)&Asl[buf][sr * 32 + sc] = ra;
    *(uint4*)&Bsl[buf][sr * 32 + sc] = rb0;
    *(uint4*)&Bsl[buf][sn1 * 32 + sc] = rb1;
  };
  gload(0);
  sstore(0);
  for (int kc = 0; kc < 64; ++kc) {
    __syncthreads();
    if (kc + 1 < 64) gload(kc + 1);  // global loads overlap compute
    compute(kc & 1);
    if (kc + 1 < 64) sstore((kc + 1) & 1);  // other buffer: no reader now
  }
#endif

  // Fused epilogue
  constexpr float kSqrtVar = 0.22360679774997896f;  // sqrt(0.05)
  float racc[2][8] = {};
#pragma unroll
  for (int mt = 0; mt < 2; ++mt) {
#pragma unroll
    for (int t = 0; t < 4; ++t) {
      const int n = n0 + wn * 64 + t * 16 + lanelo;
      const float bv = bias[n];
#pragma unroll
      for (int e = 0; e < 8; ++e) {
        const int row = m0 + wm * 32 + mt * 16 + hsel * 8 + e;
        float x = acc[mt][t][e] + bv;
        float mval = 1.0f / (1.0f + __expf(-x));
        float s = mval + kSqrtVar * eps[(size_t)row * 1024 + n];
        s = s < 0.0f ? 0.0f : (s > 1.0f ? 1.0f : s);
        float d = s - mval;
        racc[mt][e] += d * d;
        sample[(size_t)row * 1024 + n] = s;
      }
    }
  }
  // Reduce across the 16 lanes of each half (masks < 16 stay in-half)
#pragma unroll
  for (int mt = 0; mt < 2; ++mt) {
#pragma unroll
    for (int e = 0; e < 8; ++e) {
      float s = racc[mt][e];
      s += __shfl_xor(s, 8, 32);
      s += __shfl_xor(s, 4, 32);
      s += __shfl_xor(s, 2, 32);
      s += __shfl_xor(s, 1, 32);
      if (lanelo == 0) wpart[wm][mt * 16 + hsel * 8 + e][wn] = s;
    }
  }
  __syncthreads();
  if (tid < 128) {  // one thread per row of the block
    const int wmw = tid >> 5, mr = tid & 31;
    float s = wpart[wmw][mr][0] + wpart[wmw][mr][1] + wpart[wmw][mr][2] +
              wpart[wmw][mr][3];
    psum[(size_t)(m0 + tid) * 4 + blockIdx.x] = s;
  }
}

// ---------------------------------------------------------------------------
// Fold 4 partials per row into logprob; write entropy.
// ---------------------------------------------------------------------------
__global__ void k_logprob(const float* __restrict__ psum,
                          float* __restrict__ logprob,
                          float* __restrict__ entropy, int n) {
  int i = blockIdx.x * 256 + threadIdx.x;
  if (i >= n) return;
  constexpr float kHalfKTerm = -592.8218660580586f;  // 0.5*1024*(log2pi+logvar)
  float s = psum[i * 4] + psum[i * 4 + 1] + psum[i * 4 + 2] + psum[i * 4 + 3];
  logprob[i] = -10.0f * s - kHalfKTerm;  // -0.5/VAR = -10
  entropy[i] = kHalfKTerm + 512.0f;      // + 0.5*k
}

// ---------------------------------------------------------------------------
extern "C" void kernel_launch(void* const* d_in, const int* in_sizes, int n_in,
                              void* d_out, int out_size, void* d_ws,
                              size_t ws_size, hipStream_t stream) {
  (void)in_sizes; (void)n_in; (void)out_size; (void)ws_size;
  const float* encoding = (const float*)d_in[0];
  const float* mean     = (const float*)d_in[1];
  const float* enc_w1   = (const float*)d_in[2];
  const float* enc_w2   = (const float*)d_in[3];
  const float* pol_w1   = (const float*)d_in[4];
  const float* pol_b1   = (const float*)d_in[5];
  const float* pol_w2   = (const float*)d_in[6];
  const float* pol_b2   = (const float*)d_in[7];
  const float* pol_w3   = (const float*)d_in[8];
  const float* pol_b3   = (const float*)d_in[9];
  const float* comb_w   = (const float*)d_in[10];
  const float* comb_b   = (const float*)d_in[11];
  const float* eps      = (const float*)d_in[12];

  const int N = 16384;  // clients
  const int P = 1024;   // params

  // f16 workspace layout (all offsets 256B-aligned), total ~76.1 MB
  char* ws = (char*)d_ws;
  _Float16* w1t = (_Float16*)(ws + 0);        // [64,512]
  _Float16* w2t = (_Float16*)(ws + 65536);    // [1024,64]
  _Float16* p1t = (_Float16*)(ws + 196608);   // [32,1024]
  _Float16* p2t = (_Float16*)(ws + 262144);   // [32,32]
  _Float16* p3t = (_Float16*)(ws + 264192);   // [1024,32]
  _Float16* cwt = (_Float16*)(ws + 329728);   // [1024,2048]
  _Float16* h1  = (_Float16*)(ws + 4524032);  // [16384,64]
  _Float16* hA  = (_Float16*)(ws + 6621184);  // [16384,32]
  _Float16* hB  = (_Float16*)(ws + 7669760);  // [16384,32]
  _Float16* cat = (_Float16*)(ws + 8718336);  // [16384,2048]
  float*    pp  = (float*)  (ws + 75827200);  // [16384,4] partial sums

  auto cvt = [&](const float* s, _Float16* d, int R, int C) {
    int n = R * C;
    k_cvt_t<<<(n + 255) / 256, 256, 0, stream>>>(s, d, R, C);
  };
  cvt(enc_w1, w1t, 512, 64);
  cvt(enc_w2, w2t, 64, 1024);
  cvt(pol_w1, p1t, 1024, 32);
  cvt(pol_w2, p2t, 32, 32);
  cvt(pol_w3, p3t, 32, 1024);
  cvt(comb_w, cwt, 2048, 1024);

  // Stage 1: h1 = relu(encoding @ enc_w1)           [N,64], K=512
  k_gemm16x4<true, true, false><<<(N / 64) * 4 / 8, 256, 0, stream>>>(
      encoding, w1t, nullptr, h1, N, 64, 512, 512, 512, 64, 0);
  // Stage 2: cat[:, :1024] = h1 @ enc_w2            [N,1024], K=64
  k_gemm16x4<false, false, false><<<(N / 64) * 64 / 8, 256, 0, stream>>>(
      h1, w2t, nullptr, cat, N, 1024, 64, 64, 64, 2048, 0);
  // Stage 3a: hA = relu(mean @ pol_w1 + b1)         [N,32], K=1024
  k_gemm16x4<true, true, true><<<(N / 64) * 2 / 8, 256, 0, stream>>>(
      mean, p1t, pol_b1, hA, N, 32, 1024, 1024, 1024, 32, 0);
  // Stage 3b: hB = relu(hA @ pol_w2 + b2)           [N,32], K=32
  k_gemm16x4<false, true, true><<<(N / 64) * 2 / 8, 256, 0, stream>>>(
      hA, p2t, pol_b2, hB, N, 32, 32, 32, 32, 32, 0);
  // Stage 3c: cat[:, 1024:] = hB @ pol_w3 + b3      [N,1024], K=32
  k_gemm16x4<false, false, true><<<(N / 64) * 64 / 8, 256, 0, stream>>>(
      hB, p3t, pol_b3, cat, N, 1024, 32, 32, 32, 2048, 1024);

  // Final fused GEMM + sigmoid + sample + partial row sums
  float* out     = (float*)d_out;
  float* samp    = out;
  float* logprob = out + (size_t)N * P;
  float* entropy = logprob + N;
  dim3 fgrid(P / 256, N / 128);  // (n-blocks, m-blocks)
  k_final<<<fgrid, 512, 0, stream>>>(cat, cwt, comb_b, eps, samp, pp);
  k_logprob<<<N / 256, 256, 0, stream>>>(pp, logprob, entropy, N);
}
